// Qwen3MoeSparseMoeBlockNPU_36283883716951
// MI455X (gfx1250) — matrix operand order repaired and checked by reading, not compiled, not run on hardware
//
#include <hip/hip_runtime.h>
#include <hip/hip_bf16.h>

// ---------------------------------------------------------------------------
// MoE forward for MI455X (gfx1250): bf16 WMMA grouped GEMMs, fused SwiGLU,
// register-prefetch + LDS double buffering, fragment-order A tiles.
//   T=4096, H=2048, E=64, I=768, top_k=4, CAP=512.
// ---------------------------------------------------------------------------

typedef __attribute__((ext_vector_type(16))) __bf16 v16bf;
typedef __attribute__((ext_vector_type(8)))  __bf16 v8bf;
typedef __attribute__((ext_vector_type(8)))  float  v8f;

#define MOE_T   4096
#define MOE_H   2048
#define MOE_E   64
#define MOE_I   768
#define MOE_K   4
#define MOE_CAP 512
#define MOE_N   (MOE_T * MOE_K)

// ---------------------------------------------------------------------------
// 1) Gating: logits -> softmax -> stable top-4 -> renormalized weights.
// ---------------------------------------------------------------------------
__global__ __launch_bounds__(64) void moe_gate_topk(
    const float* __restrict__ x,      // [T,H]
    const float* __restrict__ gw,     // [E,H]
    int*   __restrict__ topk_idx,     // [T,4]
    float* __restrict__ topk_w)       // [T,4]
{
    __shared__ float row[MOE_H];
    __shared__ float logits[MOE_E];
    const int t = blockIdx.x;

    for (int i = threadIdx.x; i < MOE_H; i += 64)
        row[i] = x[(long)t * MOE_H + i];
    __syncthreads();

    {
        const int e = threadIdx.x;
        const float* __restrict__ w = gw + (long)e * MOE_H;
        float acc = 0.f;
        #pragma unroll 8
        for (int i = 0; i < MOE_H; ++i) acc += row[i] * w[i];
        logits[e] = acc;
    }
    __syncthreads();

    if (threadIdx.x == 0) {
        float mx = -3.4e38f;
        for (int i = 0; i < MOE_E; ++i) mx = logits[i] > mx ? logits[i] : mx;
        float sum = 0.f;
        for (int i = 0; i < MOE_E; ++i) sum += __expf(logits[i] - mx);

        unsigned long long used = 0ull;
        int   bi[MOE_K];
        float bw[MOE_K];
        float wsum = 0.f;
        for (int k = 0; k < MOE_K; ++k) {
            int best = 0; float bv = -3.4e38f;
            for (int i = 0; i < MOE_E; ++i) {
                if (((used >> i) & 1ull) == 0ull && logits[i] > bv) { bv = logits[i]; best = i; }
            }
            used |= (1ull << best);
            bi[k] = best;
            bw[k] = __expf(bv - mx) / sum;
            wsum += bw[k];
        }
        wsum += 1e-20f;
        for (int k = 0; k < MOE_K; ++k) {
            topk_idx[t * MOE_K + k] = bi[k];
            topk_w  [t * MOE_K + k] = bw[k] / wsum;
        }
    }
}

// ---------------------------------------------------------------------------
// 2) Routing: stable within-expert rank via wave32 ballot prefix.
// ---------------------------------------------------------------------------
__global__ __launch_bounds__(32) void moe_route(
    const int* __restrict__ topk_idx, // [N] flat expert id
    int* __restrict__ rows,           // [E,CAP] -> token id
    int* __restrict__ rank_buf,       // [N]     -> rank (or -1 dropped)
    int* __restrict__ counts)         // [E]
{
    const int e    = blockIdx.x;
    const int lane = threadIdx.x;
    int base = 0;
    for (int n0 = 0; n0 < MOE_N; n0 += 32) {
        const int n = n0 + lane;
        const bool m = (topk_idx[n] == e);
#if __has_builtin(__builtin_amdgcn_ballot_w32)
        unsigned mask = __builtin_amdgcn_ballot_w32(m);
#else
        unsigned mask = (unsigned)__ballot(m);
#endif
        if (m) {
            const int r = base + __popc(mask & ((1u << lane) - 1u));
            if (r < MOE_CAP) {
                rows[e * MOE_CAP + r] = n >> 2;
                rank_buf[n] = r;
            } else {
                rank_buf[n] = -1;
            }
        }
        base += __popc(mask);
    }
    if (lane == 0) counts[e] = base < MOE_CAP ? base : MOE_CAP;
}

// ---------------------------------------------------------------------------
// WMMA tiling (wave32, v_wmma_f32_16x16x32_bf16).
//   Block tile 64(M) x 64(N), K step 32.  8 waves: (wm 0..3) x (wn 0..1).
//
// A tile kept in LDS in FRAGMENT ORDER:
//   As[subtile(=m>>4)][lane][16 elems], 32 B contiguous per lane -> 2x ds_load_b128.
//   ISA A 16x32 layout: lane<16 -> M=lane, K {0..7,16..23}; lane>=16 -> K {8..15,24..31}.
//   Writer (row am, k-run krun*8): lane = (am&15) + ((krun&1)<<4), eoff = (krun>>1)*8.
//
// B tile row-major [32][B_LD]; per-lane frag = 16 bf16 down one column (stride B_LD).
//   ISA B 32x16 layout: lane<16 -> N=lane, K {0..7,16..23}; lane>=16 -> N=lane-16, K {8..15,24..31};
//   elements j run over K (same order as A).
// C: vgpr r: lane<16 -> (M=r, N=lane); lane>=16 -> (M=r+8, N=lane-16).
// ---------------------------------------------------------------------------
#define BM 64
#define BN 64
#define BKQ 32
#define A_ELEMS (4 * 32 * 16)   // 4 subtiles x 32 lanes x 16 bf16 = 2048
#define B_LD 72                 // 144 B row stride (16-B aligned rows)

__device__ inline v16bf ld_frag32(const __bf16* p) {
    const v8bf lo = *(const v8bf*)p;
    const v8bf hi = *(const v8bf*)(p + 8);
    v16bf r;
    #pragma unroll
    for (int j = 0; j < 8; ++j) { r[j] = lo[j]; r[j + 8] = hi[j]; }
    return r;
}

__device__ inline v16bf ld_bcol(const __bf16* p) {
    v16bf r;
    #pragma unroll
    for (int j = 0; j < 8; ++j) { r[j] = p[j * B_LD]; r[j + 8] = p[(16 + j) * B_LD]; }
    return r;
}

__device__ inline v8bf cvt8(float4 a, float4 b) {
    v8bf p;
    p[0] = (__bf16)a.x; p[1] = (__bf16)a.y; p[2] = (__bf16)a.z; p[3] = (__bf16)a.w;
    p[4] = (__bf16)b.x; p[5] = (__bf16)b.y; p[6] = (__bf16)b.z; p[7] = (__bf16)b.w;
    return p;
}

__device__ inline int a_swz_idx(int am, int krun) {
    const int sub  = am >> 4;
    const int lane = (am & 15) + ((krun & 1) << 4);
    const int eoff = (krun >> 1) << 3;
    return ((sub << 5) + lane) * 16 + eoff;
}

#define WMMA_BF16(a, b, c) \
    __builtin_amdgcn_wmma_f32_16x16x32_bf16(false, (a), false, (b), (short)0, (c), false, false)

// ---------------------------------------------------------------------------
// 3) Grouped GEMM 1 + fused SwiGLU (double-buffered, register prefetch):
//    act[e,m,n] = silu(g)*u,  [g|u] = gather(x)[m,:] @ w13[e]
// ---------------------------------------------------------------------------
__global__ __launch_bounds__(256) void moe_gemm1_swiglu(
    const float* __restrict__ x,      // [T,H]
    const float* __restrict__ w13,    // [E,H,2I]
    const int*   __restrict__ rows,   // [E,CAP]
    const int*   __restrict__ counts, // [E]
    float* __restrict__ act)          // [E,CAP,I]
{
    const int e  = blockIdx.z;
    const int mt = blockIdx.y;            // 0..7   (CAP/64)
    const int nt = blockIdx.x;            // 0..11  (I/64)
    const int cnt = counts[e];
    if (mt * BM >= cnt) return;

    __shared__ __bf16 As[2][A_ELEMS];
    __shared__ __bf16 Bg[2][BKQ * B_LD];
    __shared__ __bf16 Bu[2][BKQ * B_LD];

    const int tid  = threadIdx.x;
    const int wave = tid >> 5;
    const int lane = tid & 31;
    const int wm   = wave >> 1;
    const int wn   = wave & 1;

    // A loader: thread -> (row am, 8 consecutive k at krun*8)
    const int am   = tid >> 2;
    const int krun = tid & 3;
    const int gm   = mt * BM + am;
    const float* aptr = (gm < cnt) ? (x + (long)rows[e * MOE_CAP + gm] * MOE_H + krun * 8)
                                   : nullptr;
    const int aidx = a_swz_idx(am, krun);

    // B loader: thread -> (row bk, 8 consecutive n)
    const int bk = tid >> 3;
    const int bn = (tid & 7) * 8;
    const float* wbase = w13 + (long)e * MOE_H * (2 * MOE_I);
    const int ng = nt * BN;
    const float* bptr = wbase + (long)bk * (2 * MOE_I) + ng + bn;   // + kt*2I per step

    float4 aP0 = {}, aP1 = {}, gP0, gP1, uP0, uP1;

    // ---- prologue: fetch kt=0 ----
    if (aptr) { aP0 = *(const float4*)(aptr); aP1 = *(const float4*)(aptr + 4); }
    gP0 = *(const float4*)(bptr);             gP1 = *(const float4*)(bptr + 4);
    uP0 = *(const float4*)(bptr + MOE_I);     uP1 = *(const float4*)(bptr + MOE_I + 4);

    *(v8bf*)&As[0][aidx]           = cvt8(aP0, aP1);
    *(v8bf*)&Bg[0][bk * B_LD + bn] = cvt8(gP0, gP1);
    *(v8bf*)&Bu[0][bk * B_LD + bn] = cvt8(uP0, uP1);
    __syncthreads();

    v8f cg0 = {}, cg1 = {}, cu0 = {}, cu1 = {};
    const int bkRow = (lane & 15) + ((lane >> 4) << 4);
    const int fA = ((wm << 5) + lane) * 16;
    const int fB = ((lane >> 4) << 3) * B_LD + wn * 32 + (lane & 15);

    for (int kt = 0; kt < MOE_H; kt += BKQ) {
        const int  p    = (kt >> 5) & 1;
        const bool more = (kt + BKQ) < MOE_H;

        if (more) {   // register prefetch of next K tile
            const int kn = kt + BKQ;
            if (aptr) { aP0 = *(const float4*)(aptr + kn);
                        aP1 = *(const float4*)(aptr + kn + 4); }
            const float* bp = bptr + (long)kn * (2 * MOE_I);
            gP0 = *(const float4*)(bp);         gP1 = *(const float4*)(bp + 4);
            uP0 = *(const float4*)(bp + MOE_I); uP1 = *(const float4*)(bp + MOE_I + 4);
        }

        const v16bf af   = ld_frag32(&As[p][fA]);
        const v16bf bgf0 = ld_bcol(&Bg[p][fB]);
        const v16bf bgf1 = ld_bcol(&Bg[p][fB + 16]);
        const v16bf buf0 = ld_bcol(&Bu[p][fB]);
        const v16bf buf1 = ld_bcol(&Bu[p][fB + 16]);

        cg0 = WMMA_BF16(af, bgf0, cg0);
        cg1 = WMMA_BF16(af, bgf1, cg1);
        cu0 = WMMA_BF16(af, buf0, cu0);
        cu1 = WMMA_BF16(af, buf1, cu1);

        if (more) {
            *(v8bf*)&As[p ^ 1][aidx]           = cvt8(aP0, aP1);
            *(v8bf*)&Bg[p ^ 1][bk * B_LD + bn] = cvt8(gP0, gP1);
            *(v8bf*)&Bu[p ^ 1][bk * B_LD + bn] = cvt8(uP0, uP1);
        }
        __syncthreads();
    }

    // Epilogue: SwiGLU, write act (rows >= cnt inside this tile get exact 0).
    float* abase = act + (long)e * MOE_CAP * MOE_I;
    const int m0 = mt * BM + wm * 16 + ((lane >> 4) * 8);
    const int n0 = ng + wn * 32 + (lane & 15);
    #pragma unroll
    for (int r = 0; r < 8; ++r) {
        const int m = m0 + r;
        const float g0 = cg0[r], u0 = cu0[r];
        abase[(long)m * MOE_I + n0]      = (g0 / (1.f + __expf(-g0))) * u0;
        const float g1 = cg1[r], u1 = cu1[r];
        abase[(long)m * MOE_I + n0 + 16] = (g1 / (1.f + __expf(-g1))) * u1;
    }
}

// ---------------------------------------------------------------------------
// 4) Grouped GEMM 2: y[e] = act[e] @ w2[e]   (CAPxI @ IxH)
// ---------------------------------------------------------------------------
__global__ __launch_bounds__(256) void moe_gemm2(
    const float* __restrict__ act,    // [E,CAP,I]
    const float* __restrict__ w2,     // [E,I,H]
    const int*   __restrict__ counts, // [E]
    float* __restrict__ y)            // [E,CAP,H]
{
    const int e  = blockIdx.z;
    const int mt = blockIdx.y;            // 0..7  (CAP/64)
    const int nt = blockIdx.x;            // 0..31 (H/64)
    if (mt * BM >= counts[e]) return;

    __shared__ __bf16 As[2][A_ELEMS];
    __shared__ __bf16 Bs[2][BKQ * B_LD];

    const int tid  = threadIdx.x;
    const int wave = tid >> 5;
    const int lane = tid & 31;
    const int wm   = wave >> 1;
    const int wn   = wave & 1;

    const int am   = tid >> 2;
    const int krun = tid & 3;
    const float* aptr = act + ((long)e * MOE_CAP + mt * BM + am) * MOE_I + krun * 8;
    const int aidx = a_swz_idx(am, krun);

    const int bk = tid >> 3;
    const int bn = (tid & 7) * 8;
    const float* wbase = w2 + (long)e * MOE_I * MOE_H;
    const int ng = nt * BN;
    const float* bptr = wbase + (long)bk * MOE_H + ng + bn;

    float4 aP0, aP1, bP0, bP1;
    aP0 = *(const float4*)(aptr);  aP1 = *(const float4*)(aptr + 4);
    bP0 = *(const float4*)(bptr);  bP1 = *(const float4*)(bptr + 4);
    *(v8bf*)&As[0][aidx]           = cvt8(aP0, aP1);
    *(v8bf*)&Bs[0][bk * B_LD + bn] = cvt8(bP0, bP1);
    __syncthreads();

    v8f c0 = {}, c1 = {};
    const int bkRow = (lane & 15) + ((lane >> 4) << 4);
    const int fA = ((wm << 5) + lane) * 16;
    const int fB = ((lane >> 4) << 3) * B_LD + wn * 32 + (lane & 15);

    for (int kt = 0; kt < MOE_I; kt += BKQ) {
        const int  p    = (kt >> 5) & 1;
        const bool more = (kt + BKQ) < MOE_I;

        if (more) {
            const int kn = kt + BKQ;
            aP0 = *(const float4*)(aptr + kn);  aP1 = *(const float4*)(aptr + kn + 4);
            const float* bp = bptr + (long)kn * MOE_H;
            bP0 = *(const float4*)(bp);         bP1 = *(const float4*)(bp + 4);
        }

        const v16bf af = ld_frag32(&As[p][fA]);
        const v16bf b0 = ld_bcol(&Bs[p][fB]);
        const v16bf b1 = ld_bcol(&Bs[p][fB + 16]);
        c0 = WMMA_BF16(af, b0, c0);
        c1 = WMMA_BF16(af, b1, c1);

        if (more) {
            *(v8bf*)&As[p ^ 1][aidx]           = cvt8(aP0, aP1);
            *(v8bf*)&Bs[p ^ 1][bk * B_LD + bn] = cvt8(bP0, bP1);
        }
        __syncthreads();
    }

    float* ybase = y + (long)e * MOE_CAP * MOE_H;
    const int m0 = mt * BM + wm * 16 + ((lane >> 4) * 8);
    const int n0 = ng + wn * 32 + (lane & 15);
    #pragma unroll
    for (int r = 0; r < 8; ++r) {
        const int m = m0 + r;
        ybase[(long)m * MOE_H + n0]      = c0[r];
        ybase[(long)m * MOE_H + n0 + 16] = c1[r];
    }
}

// ---------------------------------------------------------------------------
// 5) Finalize: out[t] = sum_k w[t,k] * y[e_k, rank_k]
// ---------------------------------------------------------------------------
__global__ __launch_bounds__(256) void moe_finalize(
    const float* __restrict__ y,        // [E,CAP,H]
    const int*   __restrict__ topk_idx, // [T,4]
    const int*   __restrict__ rank_buf, // [T,4]
    const float* __restrict__ topk_w,   // [T,4]
    float* __restrict__ out)            // [T,H]
{
    const int t = blockIdx.x;
    int   ei[MOE_K], ri[MOE_K];
    float wi[MOE_K];
    #pragma unroll
    for (int k = 0; k < MOE_K; ++k) {
        ei[k] = topk_idx[t * MOE_K + k];
        ri[k] = rank_buf[t * MOE_K + k];
        wi[k] = topk_w  [t * MOE_K + k];
    }
    for (int h = threadIdx.x; h < MOE_H; h += 256) {
        float acc = 0.f;
        #pragma unroll
        for (int k = 0; k < MOE_K; ++k) {
            if (ri[k] >= 0)
                acc += wi[k] * y[((long)ei[k] * MOE_CAP + ri[k]) * MOE_H + h];
        }
        out[(long)t * MOE_H + h] = acc;
    }
}

// ---------------------------------------------------------------------------
// Launcher
// ---------------------------------------------------------------------------
extern "C" void kernel_launch(void* const* d_in, const int* in_sizes, int n_in,
                              void* d_out, int out_size, void* d_ws, size_t ws_size,
                              hipStream_t stream) {
    const float* x    = (const float*)d_in[0];  // hidden_states [T,H]
    const float* gw   = (const float*)d_in[1];  // gate_w        [E,H]
    const float* w13  = (const float*)d_in[2];  // w13           [E,H,2I]
    const float* w2   = (const float*)d_in[3];  // w2            [E,I,H]
    (void)in_sizes; (void)n_in; (void)out_size; (void)ws_size;   // top_k fixed at 4

    char* ws = (char*)d_ws;
    size_t off = 0;
    auto carve = [&](size_t bytes) -> char* {
        char* p = ws + off;
        off = (off + bytes + 255) & ~(size_t)255;
        return p;
    };
    float* topk_w   = (float*)carve((size_t)MOE_T * MOE_K * sizeof(float));
    int*   topk_idx = (int*)  carve((size_t)MOE_T * MOE_K * sizeof(int));
    int*   rank_buf = (int*)  carve((size_t)MOE_N * sizeof(int));
    int*   rows     = (int*)  carve((size_t)MOE_E * MOE_CAP * sizeof(int));
    int*   counts   = (int*)  carve((size_t)MOE_E * sizeof(int));
    float* act      = (float*)carve((size_t)MOE_E * MOE_CAP * MOE_I * sizeof(float)); // 100.7 MB
    float* ybuf     = (float*)carve((size_t)MOE_E * MOE_CAP * MOE_H * sizeof(float)); // 268.4 MB

    moe_gate_topk<<<MOE_T, 64, 0, stream>>>(x, gw, topk_idx, topk_w);
    moe_route<<<MOE_E, 32, 0, stream>>>(topk_idx, rows, rank_buf, counts);

    dim3 g1(MOE_I / BN, MOE_CAP / BM, MOE_E);   // 12 x 8 x 64
    moe_gemm1_swiglu<<<g1, 256, 0, stream>>>(x, w13, rows, counts, act);

    dim3 g2(MOE_H / BN, MOE_CAP / BM, MOE_E);   // 32 x 8 x 64
    moe_gemm2<<<g2, 256, 0, stream>>>(act, w2, counts, ybuf);

    moe_finalize<<<MOE_T, 256, 0, stream>>>(ybuf, topk_idx, rank_buf, topk_w, (float*)d_out);
}
